// TimesFMAttention_90890097918684
// MI455X (gfx1250) — compile-verified, matrix operand order checked by reading, not run on hardware
//
#include <hip/hip_runtime.h>
#include <hip/hip_bf16.h>

// Problem constants (match reference)
#define B_SZ   2
#define S_LEN  2048
#define HIDD   1024
#define NH     16
#define NKVH   4
#define HD     64
#define MTOT   (B_SZ * S_LEN)          // 4096 rows for all projections

typedef __attribute__((ext_vector_type(16))) _Float16 v16h;
typedef __attribute__((ext_vector_type(8)))  _Float16 v8h;
typedef __attribute__((ext_vector_type(8)))  float    v8f;

// ---------------------------------------------------------------------------
// D = A(16x32 f16) x B(32x16 f16) + C(16x16 f32), CDNA5 wave32 WMMA
// ---------------------------------------------------------------------------
__device__ __forceinline__ v8f wmma_16x16x32(v16h a, v16h b, v8f c) {
  return __builtin_amdgcn_wmma_f32_16x16x32_f16(
      /*neg_a=*/false, a, /*neg_b=*/false, b,
      /*c_mod=*/(short)0, c, /*reuse_a=*/false, /*reuse_b=*/false);
}

// Load a 16-lane-value fragment: lane holds K-chunks [c0..c0+7] and
// [c0+16..c0+23] of one row -> two contiguous 16B loads. Caller passes
// p = row_base + c0 where c0 = (lane>>4)<<3.
__device__ __forceinline__ v16h ldfrag(const _Float16* p) {
  v8h lo = *(const v8h*)(p);
  v8h hi = *(const v8h*)(p + 16);
  v16h r;
#pragma unroll
  for (int i = 0; i < 8; ++i) { r[i] = lo[i]; r[i + 8] = hi[i]; }
  return r;
}

// ---------------------------------------------------------------------------
// fp32 -> f16 conversion (one-time, for activations and weights)
// ---------------------------------------------------------------------------
__global__ void cvt_f32_f16(const float* __restrict__ x,
                            _Float16* __restrict__ y, int n) {
  int i = blockIdx.x * 256 + threadIdx.x;
  if (i < n) y[i] = (_Float16)x[i];
}

// ---------------------------------------------------------------------------
// C[M,N] = A[M,K] @ W[N,K]^T + bias, WMMA-tiled.
// Block = 256 threads = 8 waves; each wave owns a 16(M) x 64(N) output strip
// (4 accumulators), block covers 128(M) x 64(N). Grid = (M/128, N/64).
// mode 0: f16 store row-major [M,N]
// mode 1: f16 store, epilogue *= (r_softplus_0/sqrt(D)) * softplus(scaling[n&63])  (Q)
// mode 2: f16 scatter-store into Vt[b, kvh, d, s]                                   (V)
// mode 3: f32 store row-major [M,N]                                          (final O)
// ---------------------------------------------------------------------------
__global__ __launch_bounds__(256)
void gemm_xwt(const _Float16* __restrict__ A, const _Float16* __restrict__ W,
              const float* __restrict__ bias, const float* __restrict__ scaling,
              void* __restrict__ Cout, int K, int N, int mode) {
  const int tid  = threadIdx.x;
  const int lane = tid & 31;
  const int wave = tid >> 5;                 // 0..7
  const int m0   = (blockIdx.x * 8 + wave) * 16;
  const int n0   = blockIdx.y * 64;
  const int c0   = (lane >> 4) << 3;
  const int rlow = lane & 15;

  v8f acc[4] = {v8f{}, v8f{}, v8f{}, v8f{}};
  const _Float16* Arow = A + (size_t)(m0 + rlow) * K + c0;

  for (int k0 = 0; k0 < K; k0 += 32) {
    v16h af = ldfrag(Arow + k0);
#pragma unroll
    for (int j = 0; j < 4; ++j) {
      const _Float16* Wrow = W + (size_t)(n0 + j * 16 + rlow) * K + c0;
      v16h bf = ldfrag(Wrow + k0);
      acc[j] = wmma_16x16x32(af, bf, acc[j]);
    }
  }

  // Epilogue: lane holds D[m][n] with n = lane&15, m = v + 8*(lane>>4)
  const int nlane = lane & 15;
  const int mhi   = (lane >> 4) << 3;
#pragma unroll
  for (int j = 0; j < 4; ++j) {
    const int n = n0 + j * 16 + nlane;
    float colscale = 1.0f;
    if (mode == 1) {
      // (r_softplus_0 / sqrt(64)) * softplus(scaling[d]);  1.442695041/8
      float x = scaling[n & (HD - 1)];
      colscale = 0.18033688f * log1pf(__expf(x));
    }
#pragma unroll
    for (int v = 0; v < 8; ++v) {
      const int m = m0 + v + mhi;
      float val = acc[j][v] + bias[n];
      if (mode == 1) val *= colscale;
      if (mode == 2) {
        // Vt[b][kvh][d][s] <- value at (m = b*S+s, n = kvh*64+d)
        const int kvh = n >> 6, d = n & (HD - 1);
        const int b = m >> 11, s = m & (S_LEN - 1);
        ((_Float16*)Cout)[(((size_t)(b * NKVH + kvh)) * HD + d) * S_LEN + s] =
            (_Float16)val;
      } else if (mode == 3) {
        ((float*)Cout)[(size_t)m * N + n] = val;
      } else {
        ((_Float16*)Cout)[(size_t)m * N + n] = (_Float16)val;
      }
    }
  }
}

// ---------------------------------------------------------------------------
// Attention: one block (128 threads = 4 waves) per (q-tile of 16, head, batch).
// Phase 1: scores tile-by-tile via WMMA into LDS (16 x 2048 fp32) + mask.
// Phase 2: row softmax; normalized fp32 scores -> d_out, probs stay in LDS.
// Phase 3: out = P @ V via WMMA (A from LDS, B from pre-transposed Vt).
// ---------------------------------------------------------------------------
__global__ __launch_bounds__(128)
void attention_fused(const _Float16* __restrict__ Qs,   // [B*S, H*D] scaled f16
                     const _Float16* __restrict__ Kh,   // [B*S, KVH*D] f16
                     const _Float16* __restrict__ Vt,   // [B, KVH, D, S] f16
                     const float* __restrict__ mask,    // [B,1,S,S] f32
                     float* __restrict__ scores_out,    // [B,H,S,S] f32
                     _Float16* __restrict__ attn_out) { // [B*S, H*D] f16
  extern __shared__ float sc[];                 // [16][S_LEN] + red[16][8]
  float* red = sc + 16 * S_LEN;

  const int qt   = blockIdx.x;
  const int h    = blockIdx.y;
  const int b    = blockIdx.z;
  const int kvh  = h >> 2;                      // Q_PER_KV = 4
  const int q0   = qt * 16;
  const int tid  = threadIdx.x;
  const int lane = tid & 31;
  const int wave = tid >> 5;                    // 0..3
  const int c0   = (lane >> 4) << 3;
  const int rlow = lane & 15;

  // ---- Phase 1: raw scores + mask into LDS --------------------------------
  const _Float16* Qrow = Qs + ((size_t)(b * S_LEN + q0 + rlow)) * (NH * HD) + h * HD + c0;
  v16h aq0 = ldfrag(Qrow);          // d-chunk 0..31
  v16h aq1 = ldfrag(Qrow + 32);     // d-chunk 32..63

  const int nlane = lane & 15;
  const int mhi   = (lane >> 4) << 3;
  for (int kt = wave * 32; kt < wave * 32 + 32; ++kt) {
    const int k0 = kt * 16;
    const _Float16* Krow =
        Kh + ((size_t)(b * S_LEN + k0 + rlow)) * (NKVH * HD) + kvh * HD + c0;
    v16h bk0 = ldfrag(Krow);
    v16h bk1 = ldfrag(Krow + 32);
    v8f acc = v8f{};
    acc = wmma_16x16x32(aq0, bk0, acc);
    acc = wmma_16x16x32(aq1, bk1, acc);
#pragma unroll
    for (int v = 0; v < 8; ++v) {
      const int m = v + mhi;                    // local q row 0..15
      const float mk =
          mask[(((size_t)b) * S_LEN + (q0 + m)) * S_LEN + k0 + nlane];
      sc[m * S_LEN + k0 + nlane] = acc[v] + mk;
    }
  }
  __syncthreads();

  // ---- Phase 2: softmax (16 rows, 8 threads per row) ----------------------
  const int r  = tid >> 3;                      // 0..15
  const int c8 = tid & 7;                       // 0..7 -> 256-wide strips
  float mx = -3.4e38f;
  for (int c = c8 * 256; c < c8 * 256 + 256; ++c)
    mx = fmaxf(mx, sc[r * S_LEN + c]);
  red[r * 8 + c8] = mx;
  __syncthreads();
  float rmx = red[r * 8];
#pragma unroll
  for (int i = 1; i < 8; ++i) rmx = fmaxf(rmx, red[r * 8 + i]);
  __syncthreads();                              // before red is reused for sums

  float sum = 0.0f;
  for (int c = c8 * 256; c < c8 * 256 + 256; ++c) {
    float e = __expf(sc[r * S_LEN + c] - rmx);
    sc[r * S_LEN + c] = e;
    sum += e;
  }
  red[r * 8 + c8] = sum;
  __syncthreads();
  float rsum = 0.0f;
#pragma unroll
  for (int i = 0; i < 8; ++i) rsum += red[r * 8 + i];
  const float inv = 1.0f / rsum;

  float* srow = scores_out + ((((size_t)b * NH + h) * S_LEN) + q0 + r) * S_LEN;
  for (int c = c8 * 256; c < c8 * 256 + 256; ++c) {
    float p = sc[r * S_LEN + c] * inv;
    sc[r * S_LEN + c] = p;                      // keep probs in LDS for P@V
    srow[c] = p;                                // fp32 softmax output
  }
  __syncthreads();

  // ---- Phase 3: out = P @ V, wave owns 16 d-columns -----------------------
  const int n0 = wave * 16;
  const _Float16* Vrow =
      Vt + (((size_t)(b * NKVH + kvh)) * HD + n0 + rlow) * S_LEN + c0;
  v8f acc = v8f{};
  for (int k0 = 0; k0 < S_LEN; k0 += 32) {
    v16h a;
#pragma unroll
    for (int i = 0; i < 16; ++i) {
      const int kk = k0 + ((i < 8) ? (c0 + i) : (16 + c0 + (i - 8)));
      a[i] = (_Float16)sc[rlow * S_LEN + kk];
    }
    v16h bb = ldfrag(Vrow + k0);
    acc = wmma_16x16x32(a, bb, acc);
  }
#pragma unroll
  for (int v = 0; v < 8; ++v) {
    const int m = v + mhi;
    attn_out[((size_t)(b * S_LEN + q0 + m)) * (NH * HD) + h * HD + n0 + nlane] =
        (_Float16)acc[v];
  }
}

// ---------------------------------------------------------------------------
extern "C" void kernel_launch(void* const* d_in, const int* in_sizes, int n_in,
                              void* d_out, int out_size, void* d_ws, size_t ws_size,
                              hipStream_t stream) {
  const float* hidden  = (const float*)d_in[0];
  const float* mask    = (const float*)d_in[1];
  const float* scaling = (const float*)d_in[2];
  const float* Wq = (const float*)d_in[3];  const float* bq = (const float*)d_in[4];
  const float* Wk = (const float*)d_in[5];  const float* bk = (const float*)d_in[6];
  const float* Wv = (const float*)d_in[7];  const float* bv = (const float*)d_in[8];
  const float* Wo = (const float*)d_in[9];  const float* bo = (const float*)d_in[10];

  float* scores_out = (float*)d_out;                                   // [B,H,S,S]
  float* final_out  = (float*)d_out + (size_t)B_SZ * NH * S_LEN * S_LEN; // [B,S,HID]

  // f16 workspace carve-out (~34 MB)
  _Float16* ws = (_Float16*)d_ws;
  size_t off = 0;
  auto carve = [&](size_t n) { _Float16* p = ws + off; off += n; return p; };
  _Float16* Xh    = carve((size_t)MTOT * HIDD);           // hidden f16
  _Float16* Wqh   = carve((size_t)NH * HD * HIDD);
  _Float16* Wkh   = carve((size_t)NKVH * HD * HIDD);
  _Float16* Wvh   = carve((size_t)NKVH * HD * HIDD);
  _Float16* Woh   = carve((size_t)HIDD * NH * HD);
  _Float16* Qs    = carve((size_t)MTOT * NH * HD);        // scaled Q f16
  _Float16* Kh16  = carve((size_t)MTOT * NKVH * HD);
  _Float16* Vt    = carve((size_t)B_SZ * NKVH * HD * S_LEN);
  _Float16* AttnO = carve((size_t)MTOT * NH * HD);
  (void)ws_size; (void)in_sizes; (void)n_in; (void)out_size;

  // 1) fp32 -> f16 conversions
  auto cvt = [&](const float* src, _Float16* dst, int n) {
    cvt_f32_f16<<<(n + 255) / 256, 256, 0, stream>>>(src, dst, n);
  };
  cvt(hidden, Xh, MTOT * HIDD);
  cvt(Wq, Wqh, NH * HD * HIDD);
  cvt(Wk, Wkh, NKVH * HD * HIDD);
  cvt(Wv, Wvh, NKVH * HD * HIDD);
  cvt(Wo, Woh, HIDD * NH * HD);

  // 2) projections (grid = (M/128, N/64), block = 256)
  gemm_xwt<<<dim3(MTOT / 128, (NH * HD) / 64), 256, 0, stream>>>(
      Xh, Wqh, bq, scaling, Qs, HIDD, NH * HD, /*mode=*/1);
  gemm_xwt<<<dim3(MTOT / 128, (NKVH * HD) / 64), 256, 0, stream>>>(
      Xh, Wkh, bk, nullptr, Kh16, HIDD, NKVH * HD, /*mode=*/0);
  gemm_xwt<<<dim3(MTOT / 128, (NKVH * HD) / 64), 256, 0, stream>>>(
      Xh, Wvh, bv, nullptr, Vt, HIDD, NKVH * HD, /*mode=*/2);

  // 3) fused attention: grid (S/16, H, B), 128 threads, 16x2048 fp32 LDS tile
  const size_t smem = (size_t)(16 * S_LEN + 16 * 8) * sizeof(float); // 131584 B
  attention_fused<<<dim3(S_LEN / 16, NH, B_SZ), 128, smem, stream>>>(
      Qs, Kh16, Vt, mask, scores_out, AttnO);

  // 4) output projection -> fp32 d_out
  gemm_xwt<<<dim3(MTOT / 128, HIDD / 64), 256, 0, stream>>>(
      AttnO, Woh, bo, nullptr, final_out, NH * HD, HIDD, /*mode=*/3);
}
